// RQSplineCouplingLayer_84129819394189
// MI455X (gfx1250) — matrix operand order repaired and checked by reading, not compiled, not run on hardware
//
#include <hip/hip_runtime.h>
#include <hip/hip_bf16.h>

typedef __attribute__((ext_vector_type(16))) _Float16     v16h;
typedef __attribute__((ext_vector_type(8)))  float        v8f;
typedef __attribute__((ext_vector_type(4)))  unsigned int v4u;

typedef __attribute__((address_space(3))) _Float16 lds_f16_t;

union HVec { v4u u[2]; v16h h; };

// Problem constants
#define NB   32
#define HWD  128
#define NPIX (NB * HWD * HWD)          // 524288 pixels
#define MIDC 64

// Workspace layout (bytes)
#define OFF_WRB   0                          // 4 layers * 9*64*64 f16 = 294912 B
#define OFF_WOUT  294912                     // 9*80*64 f16 = 92160 B
#define OFF_STATS 387072                     // 256*(mean,rstd) f32 = 2048 B
#define OFF_PART  389120                     // 6144 f32 block partials = 24576 B
#define OFF_H     524288                     // f16 NHWC [NPIX][64] = 64 MiB
#define OFF_T     (OFF_H + 67108864)
#define OFF_U     (OFF_T + 67108864)
#define OFF_RAW   OFF_T                      // f16 [NPIX][80], aliases T+U (dead then)

// ---------------------------------------------------------------------------
// Weight repack: OIHW f32 -> [tap][o][i] f16  (rb convs 64x64, out conv 75->80 pad)
// ---------------------------------------------------------------------------
__global__ void prep_weights(const float* __restrict__ w0, const float* __restrict__ w1,
                             const float* __restrict__ w2, const float* __restrict__ w3,
                             const float* __restrict__ wo,
                             _Float16* __restrict__ wrb, _Float16* __restrict__ wout) {
  int idx = blockIdx.x * 256 + threadIdx.x;
  if (idx < 4 * 36864) {
    int l = idx / 36864, r = idx % 36864;
    int tap = r / 4096, o = (r >> 6) & 63, i = r & 63;
    const float* src = (l == 0) ? w0 : (l == 1) ? w1 : (l == 2) ? w2 : w3;
    wrb[l * 36864 + (tap * 64 + o) * 64 + i] = (_Float16)src[(o * 64 + i) * 9 + tap];
  } else {
    int r = idx - 4 * 36864;
    if (r < 9 * 80 * 64) {
      int tap = r / 5120, o = (r >> 6) % 80, i = r & 63;
      float v = (o < 75) ? wo[(o * 64 + i) * 9 + tap] : 0.f;
      wout[(tap * 80 + o) * 64 + i] = (_Float16)v;
    }
  }
}

// ---------------------------------------------------------------------------
// Input conv: h0 = relu(conv3x3(x*mask; 3->64) + b), output f16 NHWC
// ---------------------------------------------------------------------------
__global__ void conv_in(const float* __restrict__ x, const float* __restrict__ mask,
                        const float* __restrict__ wN, const float* __restrict__ bN,
                        _Float16* __restrict__ out) {
  int idx = blockIdx.x * 256 + threadIdx.x;       // p*64 + oc
  int oc = idx & 63, p = idx >> 6;
  int b = p >> 14, h = (p >> 7) & 127, w = p & 127;
  float s = bN[oc];
#pragma unroll
  for (int ky = 0; ky < 3; ++ky) {
    int hh = h + ky - 1;
    if (hh < 0 || hh >= HWD) continue;
#pragma unroll
    for (int kx = 0; kx < 3; ++kx) {
      int ww = w + kx - 1;
      if (ww < 0 || ww >= HWD) continue;
      float mv = mask[hh * HWD + ww];
#pragma unroll
      for (int c = 0; c < 3; ++c) {
        float xv = x[((b * 3 + c) << 14) + hh * HWD + ww];
        s += xv * mv * wN[((oc * 3 + c) * 3 + ky) * 3 + kx];
      }
    }
  }
  out[(p << 6) + oc] = (_Float16)fmaxf(s, 0.f);
}

// ---------------------------------------------------------------------------
// WMMA implicit-GEMM conv3x3: f16 NHWC[.,64] -> f16 NHWC[.,NOUT] (+bias)
// Each wave: M = 32 pixels (two 16-pixel A tiles, shared B) x N = NOUT,
// K = 9 taps x 64 ch (2x K32).  Weights staged per-tap into LDS via the
// CDNA5 async-DMA path (global_load_async_to_lds_b128 + s_wait_asynccnt).
// ---------------------------------------------------------------------------
template <int NOUT, int NT, int NBIAS>
__global__ void __launch_bounds__(256)
conv_wmma(const _Float16* __restrict__ in, const _Float16* __restrict__ wbase,
          const float* __restrict__ bias, _Float16* __restrict__ out) {
  __shared__ _Float16 lw[NOUT * 64];
  const int lane = threadIdx.x & 31;
  const int wave = threadIdx.x >> 5;
  const int tile = blockIdx.x * 8 + wave;         // 16384 tiles of 32 pixels
  const int p0 = tile << 5;
  const int b = p0 >> 14, hp = (p0 >> 7) & 127, w0 = p0 & 127;
  const int m   = lane & 15;                      // A row (pixel within tile)
  const int nl  = lane & 15;                      // B/D column
  const int khA = (lane < 16) ? 0 : 8;            // A-layout K half
  const int khB = (lane < 16) ? 0 : 8;            // B-layout K half

  // 32-bit LDS byte offset of lw for the async-DMA destination
  const unsigned lwoff = (unsigned)(unsigned long long)(lds_f16_t*)lw;

  v8f acc[2][NT];
  const v8f vz = {0.f, 0.f, 0.f, 0.f, 0.f, 0.f, 0.f, 0.f};
#pragma unroll
  for (int h2 = 0; h2 < 2; ++h2)
#pragma unroll
    for (int t = 0; t < NT; ++t) acc[h2][t] = vz;

  for (int tap = 0; tap < 9; ++tap) {
    __syncthreads();   // previous tap's LDS reads done before DMA overwrites
    {   // async DMA stage of this tap's weights (NOUT*8 x 16B chunks)
      const _Float16* gsrc = wbase + tap * NOUT * 64;
      for (int i = threadIdx.x; i < NOUT * 8; i += 256) {
        unsigned long long ga = (unsigned long long)gsrc + (unsigned long long)i * 16ull;
        unsigned lo = lwoff + (unsigned)i * 16u;
        asm volatile("global_load_async_to_lds_b128 %0, %1, off"
                     :: "v"(lo), "v"(ga) : "memory");
      }
      asm volatile("s_wait_asynccnt 0x0" ::: "memory");
    }
    __syncthreads();

    const int dy = tap / 3 - 1, dx = tap % 3 - 1;
    const int hh = hp + dy;
    const bool hok = (hh >= 0) && (hh < HWD);
    const _Float16* rowp = in + (((b << 7) + hh) << 7) * 64;

#pragma unroll
    for (int ck = 0; ck < 64; ck += 32) {
      HVec a[2];
#pragma unroll
      for (int h2 = 0; h2 < 2; ++h2) {
        const int wwm = w0 + h2 * 16 + m + dx;
        const bool valid = hok && (wwm >= 0) && (wwm < HWD);
        const _Float16* ap = rowp + wwm * 64;
        if (valid) {
          a[h2].u[0] = *reinterpret_cast<const v4u*>(ap + ck + khA);
          a[h2].u[1] = *reinterpret_cast<const v4u*>(ap + ck + khA + 16);
        } else {
          v4u z = {0u, 0u, 0u, 0u};
          a[h2].u[0] = z; a[h2].u[1] = z;
        }
      }
#pragma unroll
      for (int nt = 0; nt < NT; ++nt) {
        const int n = nt * 16 + nl;
        HVec bb;
        const v4u* lp = reinterpret_cast<const v4u*>(lw + (n * 64 + ck + khB));
        bb.u[0] = lp[0];
        bb.u[1] = lp[2];
        acc[0][nt] = __builtin_amdgcn_wmma_f32_16x16x32_f16(
            false, a[0].h, false, bb.h, (short)0, acc[0][nt], false, false);
        acc[1][nt] = __builtin_amdgcn_wmma_f32_16x16x32_f16(
            false, a[1].h, false, bb.h, (short)0, acc[1][nt], false, false);
      }
    }
  }

  // Epilogue: D layout -> lane holds n = nl, rows r + 8*(lane>=16)
  const int rbase = (lane < 16) ? 0 : 8;
#pragma unroll
  for (int h2 = 0; h2 < 2; ++h2) {
#pragma unroll
    for (int nt = 0; nt < NT; ++nt) {
      const int n = nt * 16 + nl;
      const float bv = (n < NBIAS) ? bias[n] : 0.f;
#pragma unroll
      for (int r = 0; r < 8; ++r) {
        const int mrow = h2 * 16 + rbase + r;
        out[((((b << 7) + hp) << 7) + (w0 + mrow)) * NOUT + n] =
            (_Float16)(acc[h2][nt][r] + bv);
      }
    }
  }
}

// ---------------------------------------------------------------------------
// GroupNorm stats: one block per (batch, group); mean & rstd over 8ch x 128x128
// ---------------------------------------------------------------------------
__global__ void __launch_bounds__(256)
gn_stats(const _Float16* __restrict__ t, float* __restrict__ stats) {
  __shared__ float rs[256], rq[256];
  const int bg = blockIdx.x, b = bg >> 3, g = bg & 7;
  float s = 0.f, q = 0.f;
  for (int i = threadIdx.x; i < 16384 * 8; i += 256) {
    int p = i >> 3, c = (g << 3) + (i & 7);
    float v = (float)t[(((b << 14) + p) << 6) + c];
    s += v; q += v * v;
  }
  rs[threadIdx.x] = s; rq[threadIdx.x] = q;
  __syncthreads();
  for (int st = 128; st > 0; st >>= 1) {
    if (threadIdx.x < st) {
      rs[threadIdx.x] += rs[threadIdx.x + st];
      rq[threadIdx.x] += rq[threadIdx.x + st];
    }
    __syncthreads();
  }
  if (threadIdx.x == 0) {
    float mean = rs[0] * (1.f / 131072.f);
    float var  = rq[0] * (1.f / 131072.f) - mean * mean;
    stats[bg * 2]     = mean;
    stats[bg * 2 + 1] = rsqrtf(var + 1e-5f);
  }
}

// MODE 1: t = relu(gn(t)) in place.  MODE 2: hbuf += gn(t) (residual add).
template <int MODE>
__global__ void gn_apply(_Float16* __restrict__ t, const float* __restrict__ sc,
                         const float* __restrict__ bi, const float* __restrict__ stats,
                         _Float16* __restrict__ hbuf) {
  int idx = blockIdx.x * 256 + threadIdx.x;       // p*64 + c
  int c = idx & 63, p = idx >> 6, b = p >> 14, g = c >> 3;
  float mean = stats[((b << 3) + g) * 2];
  float rstd = stats[((b << 3) + g) * 2 + 1];
  float v = ((float)t[idx] - mean) * rstd * sc[c] + bi[c];
  if (MODE == 1) t[idx] = (_Float16)fmaxf(v, 0.f);
  else           hbuf[idx] = (_Float16)((float)hbuf[idx] + v);
}

// ---------------------------------------------------------------------------
// RQS spline transform + deterministic per-block log-det partial sums
// ---------------------------------------------------------------------------
__device__ inline float softplusf(float v) {
  return (v > 20.f) ? v : log1pf(__expf(v));
}

__global__ void __launch_bounds__(256)
rqs_kernel(const float* __restrict__ x, const float* __restrict__ mask,
           const _Float16* __restrict__ raw, float* __restrict__ yout,
           float* __restrict__ part) {
  const int idx = blockIdx.x * 256 + threadIdx.x;  // over B*C*H*W (NCHW)
  const int b = idx / 49152;
  const int rem = idx - b * 49152;
  const int c = rem >> 14;
  const int p2 = rem & 16383;
  const int h = p2 >> 7, w = p2 & 127;
  const float mv = mask[(h << 7) + w];
  const float xv = x[idx];
  const float xt = (1.f - mv) * xv;

  const _Float16* rp = raw + (size_t)((b << 14) + p2) * 80 + c * 25;
  float Wf[8], Hf[8], Df[9];
#pragma unroll
  for (int j = 0; j < 8; ++j) Wf[j] = (float)rp[j];
#pragma unroll
  for (int j = 0; j < 8; ++j) Hf[j] = (float)rp[8 + j];
#pragma unroll
  for (int j = 0; j < 9; ++j) Df[j] = (float)rp[16 + j];

  const float tail = 5.0f;
  const float rng = 2.f * tail - 8.f * 0.001f;
  float mw = Wf[0], mh = Hf[0];
#pragma unroll
  for (int j = 1; j < 8; ++j) { mw = fmaxf(mw, Wf[j]); mh = fmaxf(mh, Hf[j]); }
  float sw = 0.f, sh = 0.f;
#pragma unroll
  for (int j = 0; j < 8; ++j) {
    Wf[j] = __expf(Wf[j] - mw); sw += Wf[j];
    Hf[j] = __expf(Hf[j] - mh); sh += Hf[j];
  }
  const float iw = rng / sw, ih = rng / sh;
#pragma unroll
  for (int j = 0; j < 8; ++j) { Wf[j] = Wf[j] * iw + 0.001f; Hf[j] = Hf[j] * ih + 0.001f; }

  const float xc = fminf(fmaxf(xt, -(tail - 1e-6f)), tail - 1e-6f);
  float cums[9], chs[9];
  cums[0] = -tail; chs[0] = -tail;
#pragma unroll
  for (int j = 0; j < 8; ++j) { cums[j + 1] = cums[j] + Wf[j]; chs[j + 1] = chs[j] + Hf[j]; }
  int bi = 0;
#pragma unroll
  for (int k = 1; k <= 8; ++k) bi += (cums[k] < xc) ? 1 : 0;
  bi = (bi > 7) ? 7 : bi;

  float bw = Wf[0], bh = Hf[0], cwlo = cums[0], chlo = chs[0], d0 = Df[0], d1 = Df[1];
#pragma unroll
  for (int j = 1; j < 8; ++j) {
    if (bi == j) { bw = Wf[j]; bh = Hf[j]; cwlo = cums[j]; chlo = chs[j]; d0 = Df[j]; d1 = Df[j + 1]; }
  }
  const float dk  = softplusf(d0) + 0.001f;
  const float dk1 = softplusf(d1) + 0.001f;
  const float s   = bh / bw;
  const float xi  = fminf(fmaxf((xc - cwlo) / bw, 0.f), 1.f);
  const float om  = 1.f - xi;
  const float num = bh * (s * xi * xi + dk * xi * om);
  const float den = s + (dk1 + dk - 2.f * s) * xi * om;
  float yv = chlo + num / den;
  const float dn = s * s * (dk1 * xi * xi + 2.f * s * xi * om + dk * om * om);
  float ljv = __logf(fmaxf(dn, 1e-8f)) - 2.f * __logf(fmaxf(fabsf(den), 1e-8f));
  const bool oob = (xt < -tail) || (xt > tail);
  yv  = oob ? xt : yv;
  ljv = oob ? 0.f : ljv;

  yout[idx] = mv * xv + (1.f - mv) * yv;

  __shared__ float red[256];
  red[threadIdx.x] = (1.f - mv) * ljv;
  __syncthreads();
  for (int st = 128; st > 0; st >>= 1) {
    if (threadIdx.x < st) red[threadIdx.x] += red[threadIdx.x + st];
    __syncthreads();
  }
  if (threadIdx.x == 0) part[blockIdx.x] = red[0];
}

__global__ void finalize_kernel(const float* __restrict__ logpx,
                                const float* __restrict__ part,
                                float* __restrict__ out_lp) {
  int b = threadIdx.x;
  if (b < NB) {
    float s = 0.f;
    for (int i = 0; i < 192; ++i) s += part[b * 192 + i];   // fixed order: deterministic
    out_lp[b] = logpx[b] + s;
  }
}

// ---------------------------------------------------------------------------
extern "C" void kernel_launch(void* const* d_in, const int* in_sizes, int n_in,
                              void* d_out, int out_size, void* d_ws, size_t ws_size,
                              hipStream_t stream) {
  (void)in_sizes; (void)n_in; (void)out_size; (void)ws_size;
  const float* x      = (const float*)d_in[0];
  const float* mask   = (const float*)d_in[1];
  const float* logpx  = (const float*)d_in[2];
  const float* cin_w  = (const float*)d_in[3];
  const float* cin_b  = (const float*)d_in[4];
  const float* rb1c1w = (const float*)d_in[5];
  const float* rb1c1b = (const float*)d_in[6];
  const float* rb1g1s = (const float*)d_in[7];
  const float* rb1g1b = (const float*)d_in[8];
  const float* rb1c2w = (const float*)d_in[9];
  const float* rb1c2b = (const float*)d_in[10];
  const float* rb1g2s = (const float*)d_in[11];
  const float* rb1g2b = (const float*)d_in[12];
  const float* rb2c1w = (const float*)d_in[13];
  const float* rb2c1b = (const float*)d_in[14];
  const float* rb2g1s = (const float*)d_in[15];
  const float* rb2g1b = (const float*)d_in[16];
  const float* rb2c2w = (const float*)d_in[17];
  const float* rb2c2b = (const float*)d_in[18];
  const float* rb2g2s = (const float*)d_in[19];
  const float* rb2g2b = (const float*)d_in[20];
  const float* cout_w = (const float*)d_in[21];
  const float* cout_b = (const float*)d_in[22];

  char* ws = (char*)d_ws;
  _Float16* WRB   = (_Float16*)(ws + OFF_WRB);
  _Float16* WOUT  = (_Float16*)(ws + OFF_WOUT);
  float*    STATS = (float*)(ws + OFF_STATS);
  float*    PART  = (float*)(ws + OFF_PART);
  _Float16* HB    = (_Float16*)(ws + OFF_H);
  _Float16* TB    = (_Float16*)(ws + OFF_T);
  _Float16* UB    = (_Float16*)(ws + OFF_U);
  _Float16* RAW   = (_Float16*)(ws + OFF_RAW);
  float* yout   = (float*)d_out;
  float* out_lp = yout + (size_t)NB * 3 * HWD * HWD;

  prep_weights<<<756, 256, 0, stream>>>(rb1c1w, rb1c2w, rb2c1w, rb2c2w, cout_w, WRB, WOUT);
  conv_in<<<131072, 256, 0, stream>>>(x, mask, cin_w, cin_b, HB);

  // residual block 1
  conv_wmma<64, 4, 64><<<2048, 256, 0, stream>>>(HB, WRB + 0 * 36864, rb1c1b, TB);
  gn_stats<<<256, 256, 0, stream>>>(TB, STATS);
  gn_apply<1><<<131072, 256, 0, stream>>>(TB, rb1g1s, rb1g1b, STATS, nullptr);
  conv_wmma<64, 4, 64><<<2048, 256, 0, stream>>>(TB, WRB + 1 * 36864, rb1c2b, UB);
  gn_stats<<<256, 256, 0, stream>>>(UB, STATS);
  gn_apply<2><<<131072, 256, 0, stream>>>(UB, rb1g2s, rb1g2b, STATS, HB);

  // residual block 2
  conv_wmma<64, 4, 64><<<2048, 256, 0, stream>>>(HB, WRB + 2 * 36864, rb2c1b, TB);
  gn_stats<<<256, 256, 0, stream>>>(TB, STATS);
  gn_apply<1><<<131072, 256, 0, stream>>>(TB, rb2g1s, rb2g1b, STATS, nullptr);
  conv_wmma<64, 4, 64><<<2048, 256, 0, stream>>>(TB, WRB + 3 * 36864, rb2c2b, UB);
  gn_stats<<<256, 256, 0, stream>>>(UB, STATS);
  gn_apply<2><<<131072, 256, 0, stream>>>(UB, rb2g2s, rb2g2b, STATS, HB);

  // output conv 64 -> 75 (padded to 80); RAW aliases TB/UB which are dead here
  conv_wmma<80, 5, 75><<<2048, 256, 0, stream>>>(HB, WOUT, cout_b, RAW);

  // spline + per-block logdet partials, then deterministic finalize
  rqs_kernel<<<6144, 256, 0, stream>>>(x, mask, RAW, yout, PART);
  finalize_kernel<<<1, 32, 0, stream>>>(logpx, PART, out_lp);
}
